// DCGRUDecoder_10273561772735
// MI455X (gfx1250) — matrix operand order repaired and checked by reading, not compiled, not run on hardware
//
#include <hip/hip_runtime.h>
#include <hip/hip_bf16.h>
#include <math.h>

// ---------------- problem constants ----------------
#define SEQ 32
#define BB  1024
#define NN  19
#define HH  128
#define MM  (BB*NN)          // 19456 rows = 1216 * 16
#define KP0 416              // ceil(3*129 / 32)*32
#define KP1 768              // 3*256

typedef __attribute__((ext_vector_type(16))) __bf16 v16bf;
typedef __attribute__((ext_vector_type(8)))  __bf16 v8bf;
typedef __attribute__((ext_vector_type(8)))  float  v8f;

// ---------------- helpers ----------------
__global__ void copy_f32(const float* __restrict__ src, float* __restrict__ dst, int n) {
    int i = blockIdx.x * 256 + threadIdx.x;
    if (i < n) dst[i] = src[i];
}

// Convert fp32 weight (Krows x Nout, row-major) to bf16 column-major padded:
// Wt[n*K_pad + k] = (k < Krows) ? W[k*Nout + n] : 0
__global__ void wconv_kernel(const float* __restrict__ W, __bf16* __restrict__ Wt,
                             int Krows, int K_pad, int Nout) {
    int tid = blockIdx.x * 256 + threadIdx.x;
    if (tid >= Nout * K_pad) return;
    int n = tid / K_pad;
    int k = tid - n * K_pad;
    Wt[(size_t)n * K_pad + k] = (k < Krows) ? (__bf16)W[(size_t)k * Nout + n] : (__bf16)0.0f;
}

// ---------------- Chebyshev graph-diffusion feature build ----------------
// Builds X (MM x K_pad, bf16, row-major) = [x0 | S@x0 | 2S@(S@x0)-x0] (+ zero pad cols)
__global__ void cheb_build(const float* __restrict__ xin, const float* __restrict__ h,
                           const float* __restrict__ ru,  const float* __restrict__ S,
                           __bf16* __restrict__ X, int F1, int F, int Fp, int K_pad) {
    __shared__ float sS[NN * NN];
    for (int i = threadIdx.x; i < NN * NN; i += 256) sS[i] = S[i];
    __syncthreads();

    int tid = blockIdx.x * 256 + threadIdx.x;
    int b = tid / Fp;
    int f = tid - b * Fp;
    if (b >= BB) return;
    size_t rowb = (size_t)b * NN;

    if (f >= F) {                       // zero the K padding columns
        int p = 3 * F + (f - F);
        for (int m = 0; m < NN; ++m) X[(rowb + m) * K_pad + p] = (__bf16)0.0f;
        return;
    }

    float x0[NN], x1[NN];
    if (f < F1) {
        #pragma unroll
        for (int n = 0; n < NN; ++n) x0[n] = xin[(rowb + n) * F1 + f];
    } else {
        int hc = f - F1;
        if (ru) {
            #pragma unroll
            for (int n = 0; n < NN; ++n)
                x0[n] = h[(rowb + n) * HH + hc] * ru[(rowb + n) * 256 + hc];
        } else {
            #pragma unroll
            for (int n = 0; n < NN; ++n) x0[n] = h[(rowb + n) * HH + hc];
        }
    }
    #pragma unroll
    for (int m = 0; m < NN; ++m) {
        float s = 0.f;
        #pragma unroll
        for (int n = 0; n < NN; ++n) s = fmaf(sS[m * NN + n], x0[n], s);
        x1[m] = s;
    }
    #pragma unroll
    for (int m = 0; m < NN; ++m) {
        float s = 0.f;
        #pragma unroll
        for (int n = 0; n < NN; ++n) s = fmaf(sS[m * NN + n], x1[n], s);
        float x2 = 2.f * s - x0[m];
        size_t base = (rowb + m) * K_pad;
        X[base + f]         = (__bf16)x0[m];
        X[base + F + f]     = (__bf16)x1[m];
        X[base + 2 * F + f] = (__bf16)x2;
    }
}

// ---------------- WMMA GEMM with LDS-staged B, fused bias + activation ----------------
// X  : MM x K_pad bf16 row-major (A), streamed from global (VMEM pipe)
// Wt : Nout x K_pad bf16 column-major; the block's 64-column panel is staged once
//      into LDS (96 KB max of the WGP's 320 KB) and served via the DS pipe.
// out: MM x Nout f32; act: 1=sigmoid, 2=tanh
// Each wave: 16(M) x 64(N) strip; block of 8 waves: 128(M) x 64(N).
__global__ __launch_bounds__(256) void gemm_bias_act(
        const __bf16* __restrict__ X, const __bf16* __restrict__ Wt,
        const float* __restrict__ bias, float* __restrict__ out,
        int K_pad, int Nout, int act) {
    __shared__ __bf16 sB[64 * KP1];    // max panel: 64 x 768 bf16 = 96 KB

    const int lane = threadIdx.x & 31;
    const int wave = threadIdx.x >> 5;
    const int r    = lane & 15;        // row-in-tile (A) / col-in-tile (B,D)
    const int half = lane >> 4;        // K-slice selector
    const int rowBase = (blockIdx.x * 8 + wave) * 16;
    const int nBase   = blockIdx.y * 64;

    // cooperative stage of the contiguous 64-row B panel (vectorized 16B copies)
    {
        const v8bf* src = (const v8bf*)(Wt + (size_t)nBase * K_pad);
        v8bf* dst = (v8bf*)sB;
        const int nvec = 64 * K_pad / 8;
        for (int i = threadIdx.x; i < nvec; i += 256) dst[i] = src[i];
    }
    __syncthreads();

    v8f acc[4];
    #pragma unroll
    for (int j = 0; j < 4; ++j)
        #pragma unroll
        for (int i = 0; i < 8; ++i) acc[j][i] = 0.0f;

    const __bf16* Arow = X + (size_t)(rowBase + r) * K_pad;
    const __bf16* sBlane = sB + half * 16;   // per-lane K-slice base within a column

    for (int k0 = 0; k0 < K_pad; k0 += 32) {
        // A fragment: contiguous K order — half 0: K{0..15}, half 1: K{16..31}
        const __bf16* ap = Arow + k0 + half * 16;
        v8bf alo = *(const v8bf*)(ap);
        v8bf ahi = *(const v8bf*)(ap + 8);
        v16bf a;
        #pragma unroll
        for (int i = 0; i < 8; ++i) { a[i] = alo[i]; a[i + 8] = ahi[i]; }

        #pragma unroll
        for (int j = 0; j < 4; ++j) {
            // B fragment from LDS: column n = j*16 + r, K-slice half*16..+15 (contiguous)
            v16bf bfrag = *(const v16bf*)(sBlane + (size_t)(j * 16 + r) * K_pad + k0);
            acc[j] = __builtin_amdgcn_wmma_f32_16x16x32_bf16(
                         false, a, false, bfrag, (short)0, acc[j], false, false);
        }
    }

    // epilogue: D(m,n) -> lane (n = lane&15), VGPR v (m = v + 8*half)
    #pragma unroll
    for (int j = 0; j < 4; ++j) {
        int n  = nBase + j * 16 + r;
        float bv = bias[n];
        #pragma unroll
        for (int v = 0; v < 8; ++v) {
            int row = rowBase + v + 8 * half;
            float val = acc[j][v] + bv;
            val = (act == 1) ? (1.0f / (1.0f + __expf(-val))) : tanhf(val);
            out[(size_t)row * Nout + n] = val;
        }
    }
}

// ---------------- GRU combine: h = u*h + (1-u)*c ----------------
__global__ void combine_kernel(const float* __restrict__ ru, const float* __restrict__ c,
                               float* __restrict__ h) {
    int tid = blockIdx.x * 256 + threadIdx.x;
    if (tid >= MM * HH) return;
    int row = tid / HH;
    int hc  = tid - row * HH;
    float u  = ru[(size_t)row * 256 + HH + hc];
    float hv = h[tid];
    float cv = c[tid];
    h[tid] = u * hv + (1.0f - u) * cv;
}

// ---------------- projection: out = h1 @ W_proj + b_proj ----------------
__global__ void proj_kernel(const float* __restrict__ h1, const float* __restrict__ Wp,
                            const float* __restrict__ bp, float* __restrict__ out) {
    int row = blockIdx.x * 256 + threadIdx.x;
    if (row >= MM) return;
    float s = bp[0];
    const float* hr = h1 + (size_t)row * HH;
    #pragma unroll 8
    for (int i = 0; i < HH; ++i) s = fmaf(hr[i], Wp[i], s);
    out[row] = s;
}

// ---------------- host orchestration ----------------
extern "C" void kernel_launch(void* const* d_in, const int* in_sizes, int n_in,
                              void* d_out, int out_size, void* d_ws, size_t ws_size,
                              hipStream_t stream) {
    const float* inputs  = (const float*)d_in[0];   // (32,1024,19,1)
    const float* h0init  = (const float*)d_in[1];   // (2,1024,2432)
    const float* support = (const float*)d_in[2];   // (19,19)
    const float* Wg0 = (const float*)d_in[3];  const float* bg0 = (const float*)d_in[4];
    const float* Wc0 = (const float*)d_in[5];  const float* bc0 = (const float*)d_in[6];
    const float* Wg1 = (const float*)d_in[7];  const float* bg1 = (const float*)d_in[8];
    const float* Wc1 = (const float*)d_in[9];  const float* bc1 = (const float*)d_in[10];
    const float* Wp  = (const float*)d_in[11]; const float* bp  = (const float*)d_in[12];
    float* out = (float*)d_out;

    // workspace carve (256B aligned)
    char* p = (char*)d_ws;
    auto carve = [&](size_t bytes) { char* q = p; p += (bytes + 255) & ~(size_t)255; return q; };
    __bf16* Xbuf = (__bf16*)carve((size_t)MM * KP1 * 2);   // shared across all gconvs
    __bf16* Wg0T = (__bf16*)carve((size_t)256 * KP0 * 2);
    __bf16* Wc0T = (__bf16*)carve((size_t)128 * KP0 * 2);
    __bf16* Wg1T = (__bf16*)carve((size_t)256 * KP1 * 2);
    __bf16* Wc1T = (__bf16*)carve((size_t)128 * KP1 * 2);
    float*  ru   = (float*) carve((size_t)MM * 256 * 4);
    float*  cbuf = (float*) carve((size_t)MM * 128 * 4);
    float*  hbuf = (float*) carve((size_t)2 * MM * 128 * 4);

    float* h0 = hbuf;
    float* h1 = hbuf + (size_t)MM * HH;

    // init hidden state (copy: inputs must not be mutated)
    {
        int n = 2 * MM * HH;
        copy_f32<<<(n + 255) / 256, 256, 0, stream>>>(h0init, hbuf, n);
    }
    // weight conversion to bf16 column-major padded
    wconv_kernel<<<(256 * KP0 + 255) / 256, 256, 0, stream>>>(Wg0, Wg0T, 387, KP0, 256);
    wconv_kernel<<<(128 * KP0 + 255) / 256, 256, 0, stream>>>(Wc0, Wc0T, 387, KP0, 128);
    wconv_kernel<<<(256 * KP1 + 255) / 256, 256, 0, stream>>>(Wg1, Wg1T, 768, KP1, 256);
    wconv_kernel<<<(128 * KP1 + 255) / 256, 256, 0, stream>>>(Wc1, Wc1T, 768, KP1, 128);

    const dim3 gemmBlk(256);
    const dim3 gGate(MM / 128, 256 / 64);   // (152, 4)
    const dim3 gCand(MM / 128, 128 / 64);   // (152, 2)
    const int  cheb0Thr = BB * (129 + (KP0 - 3 * 129));   // Fp = 158
    const int  cheb1Thr = BB * 256;
    const int  combBlks = (MM * HH + 255) / 256;

    for (int t = 0; t < SEQ; ++t) {
        const float* xin = inputs + (size_t)t * MM;   // OUT=1 feature

        // ---- layer 0 cell (F1=1, F=129, K_pad=416) ----
        cheb_build<<<(cheb0Thr + 255) / 256, 256, 0, stream>>>(
            xin, h0, nullptr, support, Xbuf, 1, 129, 158, KP0);
        gemm_bias_act<<<gGate, gemmBlk, 0, stream>>>(Xbuf, Wg0T, bg0, ru, KP0, 256, 1);
        cheb_build<<<(cheb0Thr + 255) / 256, 256, 0, stream>>>(
            xin, h0, ru, support, Xbuf, 1, 129, 158, KP0);
        gemm_bias_act<<<gCand, gemmBlk, 0, stream>>>(Xbuf, Wc0T, bc0, cbuf, KP0, 128, 2);
        combine_kernel<<<combBlks, 256, 0, stream>>>(ru, cbuf, h0);

        // ---- layer 1 cell (F1=128, F=256, K_pad=768, x_in = new h0) ----
        cheb_build<<<(cheb1Thr + 255) / 256, 256, 0, stream>>>(
            h0, h1, nullptr, support, Xbuf, 128, 256, 256, KP1);
        gemm_bias_act<<<gGate, gemmBlk, 0, stream>>>(Xbuf, Wg1T, bg1, ru, KP1, 256, 1);
        cheb_build<<<(cheb1Thr + 255) / 256, 256, 0, stream>>>(
            h0, h1, ru, support, Xbuf, 128, 256, 256, KP1);
        gemm_bias_act<<<gCand, gemmBlk, 0, stream>>>(Xbuf, Wc1T, bc1, cbuf, KP1, 128, 2);
        combine_kernel<<<combBlks, 256, 0, stream>>>(ru, cbuf, h1);

        // ---- projection ----
        proj_kernel<<<(MM + 255) / 256, 256, 0, stream>>>(h1, Wp, bp, out + (size_t)t * MM);
    }
}
